// FlowGuidedPCDAlign_81990925681200
// MI455X (gfx1250) — matrix operand order repaired and checked by reading, not compiled, not run on hardware
//
#include <hip/hip_runtime.h>
#include <hip/hip_bf16.h>

// ---------------------------------------------------------------------------
// FlowGuidedPCDAlign on gfx1250 (MI455X):
//  - all GEMM-shaped work on v_wmma_f32_16x16x32_bf16 (f32 accumulate)
//  - halo staging via global_load_async_to_lds_b128 (ASYNCcnt) from a
//    zero-padded NHWC activation layout (no bounds checks, 16B/lane)
//  - deformable bilinear sampling on VALU, f32 coordinates
// ---------------------------------------------------------------------------

#define NF   64
#define DG   8
#define KKT  9          // 3x3 taps
#define BB   2
#define HH   192
#define WW   320
#define HP   (HH + 2)   // padded height
#define WP   (WW + 2)   // padded width

typedef __attribute__((ext_vector_type(16))) __bf16 v16bf;
typedef __attribute__((ext_vector_type(8)))  float  v8f;

union BF16x16 {
    v16bf v;
    unsigned short u[16];
    uint4 q[2];
};

union BF16x8 {
    unsigned short u[8];
    uint4 q;
};

static __device__ __forceinline__ unsigned short f2bf(float f) {
    unsigned int u = __float_as_uint(f);
    unsigned int r = u + 0x7FFFu + ((u >> 16) & 1u);   // round-to-nearest-even
    return (unsigned short)(r >> 16);
}

// Load a 16-half fragment (32B) as two 16B vectors (ds_load_b128 / global b128).
static __device__ __forceinline__ v16bf frag_from(const unsigned short* p) {
    BF16x16 t;
    const uint4* q = (const uint4*)p;
    t.q[0] = q[0];
    t.q[1] = q[1];
    return t.v;
}

// Async copy of one 16B chunk global -> LDS (CDNA5 GLOBAL_LOAD_ASYNC_TO_LDS_B128).
// Low 32 bits of a generic shared-aperture pointer are the LDS byte offset.
static __device__ __forceinline__ void async_g2l_b128(const unsigned short* gp,
                                                      unsigned short* lp) {
    unsigned int laddr = (unsigned int)(size_t)lp;
    asm volatile("global_load_async_to_lds_b128 %0, %1, off"
                 :: "v"(laddr), "v"(gp) : "memory");
}

static __device__ __forceinline__ void async_wait0() {
    asm volatile("s_wait_asynccnt 0" ::: "memory");
}

// ---------------------------------------------------------------------------
// Weight packing: A-matrix fragment layout for V_WMMA_F32_16X16X32_BF16.
// Lane L: m = L%16 (= output channel within tile), hi = L/16.
// Half i: K = i + 16*hi (contiguous, matches B fragment reads)
// dst layout: [((osub*9 + tap)*KC + kc)*512 + lane*16 + i]
// ---------------------------------------------------------------------------
__global__ void pack_conv_w(const float* __restrict__ w, unsigned short* __restrict__ dst,
                            int Cout, int Cin, int KC, int OT) {
    int idx = blockIdx.x * blockDim.x + threadIdx.x;
    int total = OT * 9 * KC * 512;
    if (idx >= total) return;
    int i    = idx & 15;
    int lane = (idx >> 4) & 31;
    int tile = idx >> 9;
    int kc   = tile % KC;
    int tap  = (tile / KC) % 9;
    int osub = tile / (KC * 9);
    int m = lane & 15, hi = lane >> 4;
    int v = i >> 1, p = i & 1;
    int K = 2 * v + p + 16 * hi;
    int cin = kc * 32 + K;
    int och = osub * 16 + m;
    float val = 0.0f;
    if (cin < Cin && och < Cout)
        val = w[(((size_t)och * Cin + cin) * 3 + tap / 3) * 3 + tap % 3];
    dst[idx] = f2bf(val);
}

// DCN einsum weights: K = g*72 + k*8 + c  (g=group, k=tap, c=channel-in-group)
// dst layout: [(osub*18 + kc)*512 + lane*16 + i]
__global__ void pack_dcn_w(const float* __restrict__ w, unsigned short* __restrict__ dst) {
    int idx = blockIdx.x * blockDim.x + threadIdx.x;
    int total = 4 * 18 * 512;
    if (idx >= total) return;
    int i    = idx & 15;
    int lane = (idx >> 4) & 31;
    int tile = idx >> 9;
    int kc   = tile % 18;
    int osub = tile / 18;
    int m = lane & 15, hi = lane >> 4;
    int v = i >> 1, p = i & 1;
    int K = 2 * v + p + 16 * hi;
    int cinK = kc * 32 + K;                 // 0..575
    int g = cinK / 72, rem = cinK % 72;
    int k = rem / 8,   c = rem & 7;
    int och = osub * 16 + m;
    int cin = g * 8 + c;
    float val = w[(((size_t)och * NF + cin) * 3 + k / 3) * 3 + k % 3];
    dst[idx] = f2bf(val);
}

// Zero the 1-pixel border of a padded NHWC bf16 buffer [B][HP][WP][Cp].
__global__ void zero_border(unsigned short* __restrict__ dst, int Cp) {
    int idx = blockIdx.x * blockDim.x + threadIdx.x;
    int total = BB * HP * WP;
    if (idx >= total) return;
    int xy = idx % (HP * WP);
    int yy = xy / WP, xx = xy % WP;
    if (yy >= 1 && yy <= HH && xx >= 1 && xx <= WW) return;   // interior
    uint4 z = make_uint4(0u, 0u, 0u, 0u);
    uint4* p = (uint4*)(dst + (size_t)idx * Cp);
    for (int i = 0; i < (Cp >> 3); ++i) p[i] = z;
}

// concat(warped 64, ref 64, flows 2, zeros 30) -> padded NHWC bf16 [B][HP][WP][160]
__global__ void pack_in0(const float* __restrict__ warped, const float* __restrict__ ref,
                         const float* __restrict__ flows, unsigned short* __restrict__ dst) {
    int idx = blockIdx.x * blockDim.x + threadIdx.x;   // over B*H*W*20
    int total = BB * HH * WW * 20;
    if (idx >= total) return;
    int c8  = idx % 20;
    int pix = idx / 20;
    int b  = pix / (HH * WW);
    int yx = pix % (HH * WW);
    int y = yx / WW, xc = yx % WW;
    int c0 = c8 * 8;
    BF16x8 o;
#pragma unroll
    for (int i = 0; i < 8; ++i) {
        int c = c0 + i;
        float v = 0.0f;
        if (c < 64)       v = warped[((size_t)b * 64 + c) * HH * WW + yx];
        else if (c < 128) v = ref[((size_t)b * 64 + (c - 64)) * HH * WW + yx];
        else if (c < 130) v = flows[((size_t)b * 2 + (c - 128)) * HH * WW + yx];
        o.u[i] = f2bf(v);
    }
    *(uint4*)&dst[((((size_t)b * HP + y + 1) * WP) + xc + 1) * 160 + c0] = o.q;
}

// ---------------------------------------------------------------------------
// Generic 3x3 conv, implicit GEMM on WMMA bf16.
// in:  padded NHWC bf16 [B][HP][WP][CinP] (zero border, CinP multiple of 32)
// out: padded NHWC bf16 [B][HP][WP][CoutP] (+bias, optional leaky-ReLU 0.1)
// Block: 256 threads = 8 waves, tile = 32 pixels of one row.
// ---------------------------------------------------------------------------
__global__ void conv3_wmma(const unsigned short* __restrict__ in,
                           const unsigned short* __restrict__ wpk,
                           const float* __restrict__ bias,
                           unsigned short* __restrict__ out,
                           int CinP, int KC, int CoutP, int Cout, int lrelu) {
    extern __shared__ __attribute__((aligned(32))) unsigned short lds[]; // [3][34][CinP]
    const int tid  = threadIdx.x;
    const int lane = tid & 31;
    const int wave = tid >> 5;
    int blk = blockIdx.x;
    int xt = blk % (WW / 32);
    int y  = (blk / (WW / 32)) % HH;
    int b  = blk / ((WW / 32) * HH);
    int x0 = xt * 32;

    // Stage halo tile via async b128 copies: rows are contiguous in the padded
    // layout (34*CinP halfs each), no bounds checks needed.
    const int rowhalfs = 34 * CinP;
    const int cpr = rowhalfs >> 3;           // 16B chunks per row
    for (int ch = tid; ch < 3 * cpr; ch += blockDim.x) {
        int row = ch / cpr;
        int cc  = ch % cpr;
        const unsigned short* gp =
            in + ((((size_t)b * HP + (y + row)) * WP) + x0) * CinP + cc * 8;
        async_g2l_b128(gp, &lds[row * rowhalfs + cc * 8]);
    }
    async_wait0();
    __syncthreads();

    const int n = lane & 15, hi = lane >> 4;
    const int OT = CoutP >> 4;
    const int ntiles = 2 * OT;                 // 2 pixel-subtiles x OT och-subtiles
    const int nw = blockDim.x >> 5;
    for (int t = wave; t < ntiles; t += nw) {
        int psub = t & 1, osub = t >> 1;
        v8f acc = {};
        for (int tap = 0; tap < 9; ++tap) {
            int ty = tap / 3, tx = tap % 3;
            const unsigned short* arow = wpk + ((size_t)(osub * 9 + tap) * KC) * 512;
            int lb = (ty * 34 + psub * 16 + n + tx) * CinP;   // B: N=lane%16, K=i+16*hi
            for (int kc = 0; kc < KC; ++kc) {
                v16bf a  = frag_from(arow + kc * 512 + lane * 16);
                v16bf bf = frag_from(&lds[lb + kc * 32 + hi * 16]);
                acc = __builtin_amdgcn_wmma_f32_16x16x32_bf16(false, a, false, bf,
                                                              (short)0, acc, false, false);
            }
        }
        // C layout: N = lane%16 (pixel), M = 8*hi + r (och within tile)
        int px = x0 + psub * 16 + n;
        size_t obase = (((size_t)b * HP + y + 1) * WP + px + 1) * CoutP + osub * 16 + 8 * hi;
        BF16x8 o;
#pragma unroll
        for (int r = 0; r < 8; ++r) {
            int och = osub * 16 + 8 * hi + r;
            float v = acc[r] + ((och < Cout) ? bias[och] : 0.0f);
            if (lrelu) v = (v >= 0.0f) ? v : 0.1f * v;
            o.u[r] = f2bf(v);
        }
        *(uint4*)&out[obase] = o.q;
    }
}

// ---------------------------------------------------------------------------
// Fused: offset/mask conv (64->216, WMMA) + flow-guided deformable sampling
// + 576->64 einsum (WMMA) + bias + leaky-ReLU.  Block = 128 threads, 16 pixels.
// ---------------------------------------------------------------------------
__global__ void dcn_wmma(const unsigned short* __restrict__ off2,  // padded NHWC bf16 [B][HP][WP][64]
                         const unsigned short* __restrict__ wpkom, // [14][9][2][512]
                         const float* __restrict__ bom,            // [216]
                         const unsigned short* __restrict__ wpkd,  // [4][18][512]
                         const float* __restrict__ bd,             // [64]
                         const float* __restrict__ x,              // nbr_fea NCHW f32
                         const float* __restrict__ flows,          // NCHW f32 [B][2][H][W]
                         float* __restrict__ out) {                // NCHW f32 [B][64][H][W]
    __shared__ __attribute__((aligned(32))) unsigned short s_off2[3 * 18 * 64]; // halo tile
    __shared__ __attribute__((aligned(32))) float          s_om[16 * 224];      // om, f32
    __shared__ __attribute__((aligned(32))) unsigned short s_samp[16 * 576];    // B matrix

    const int tid  = threadIdx.x;            // 128 threads = 4 waves
    const int lane = tid & 31;
    const int wave = tid >> 5;
    int blk = blockIdx.x;
    int xt = blk % (WW / 16);
    int y  = (blk / (WW / 16)) % HH;
    int b  = blk / ((WW / 16) * HH);
    int x0 = xt * 16;

    // ---- Phase 1: async-stage off2 halo (3 rows x 18 px x 64 ch, contiguous) ----
    for (int ch = tid; ch < 3 * 144; ch += 128) {   // 18*64/8 = 144 chunks per row
        int row = ch / 144;
        int cc  = ch % 144;
        const unsigned short* gp =
            off2 + ((((size_t)b * HP + (y + row)) * WP) + x0) * 64 + cc * 8;
        async_g2l_b128(gp, &s_off2[row * 1152 + cc * 8]);
    }
    async_wait0();
    __syncthreads();

    // ---- Phase 2: om = conv3(off2) (+bias), 216 ch padded to 224, into LDS f32 ----
    const int n = lane & 15, hi = lane >> 4;
    for (int osub = wave; osub < 14; osub += 4) {
        v8f acc = {};
        for (int tap = 0; tap < 9; ++tap) {
            int ty = tap / 3, tx = tap % 3;
            const unsigned short* arow = wpkom + (size_t)(osub * 9 + tap) * 2 * 512;
            int lb = (ty * 18 + n + tx) * 64;
            for (int kc = 0; kc < 2; ++kc) {
                v16bf a  = frag_from(arow + kc * 512 + lane * 16);
                v16bf bf = frag_from(&s_off2[lb + kc * 32 + hi * 16]);
                acc = __builtin_amdgcn_wmma_f32_16x16x32_bf16(false, a, false, bf,
                                                              (short)0, acc, false, false);
            }
        }
        v8f vv;
#pragma unroll
        for (int r = 0; r < 8; ++r) {
            int och = osub * 16 + 8 * hi + r;
            vv[r] = acc[r] + ((och < 216) ? bom[och] : 0.0f);
        }
        *(v8f*)&s_om[n * 224 + osub * 16 + 8 * hi] = vv;   // 32B aligned
    }
    __syncthreads();

    // ---- Phase 3: flow-guided bilinear sampling -> bf16 B-matrix in LDS ----
    // off_y(g,k)=om[g*18+2k]+flow1, off_x=om[g*18+2k+1]+flow0 (flows[:, ::-1])
    // mask(g,k)=sigmoid(om[144+g*9+k]);  K index = g*72 + k*8 + c
    for (int item = tid; item < 16 * 72; item += 128) {
        int pix = item / 72;
        int gk  = item % 72;
        int g = gk / 9, k = gk % 9;
        int xx = x0 + pix;
        const float* omp = &s_om[pix * 224];
        float f0 = flows[(((size_t)b * 2 + 0) * HH + y) * WW + xx];
        float f1 = flows[(((size_t)b * 2 + 1) * HH + y) * WW + xx];
        float offy = omp[g * 18 + 2 * k]     + f1;
        float offx = omp[g * 18 + 2 * k + 1] + f0;
        float mval = 1.0f / (1.0f + __expf(-omp[144 + g * 9 + k]));
        float py = (float)y  + (float)(k / 3 - 1) + offy;
        float px = (float)xx + (float)(k % 3 - 1) + offx;
        float y0f = floorf(py), x0f = floorf(px);
        float wy = py - y0f, wx = px - x0f;
        int iy = (int)y0f, ix = (int)x0f;
        bool vy0 = (iy >= 0) && (iy < HH);
        bool vy1 = (iy + 1 >= 0) && (iy + 1 < HH);
        bool vx0 = (ix >= 0) && (ix < WW);
        bool vx1 = (ix + 1 >= 0) && (ix + 1 < WW);
        float c00 = (vy0 && vx0) ? (1.0f - wy) * (1.0f - wx) : 0.0f;
        float c01 = (vy0 && vx1) ? (1.0f - wy) * wx          : 0.0f;
        float c10 = (vy1 && vx0) ? wy * (1.0f - wx)          : 0.0f;
        float c11 = (vy1 && vx1) ? wy * wx                   : 0.0f;
        int iy0c = min(max(iy, 0), HH - 1),     iy1c = min(max(iy + 1, 0), HH - 1);
        int ix0c = min(max(ix, 0), WW - 1),     ix1c = min(max(ix + 1, 0), WW - 1);
        size_t i00 = (size_t)iy0c * WW + ix0c, i01 = (size_t)iy0c * WW + ix1c;
        size_t i10 = (size_t)iy1c * WW + ix0c, i11 = (size_t)iy1c * WW + ix1c;
        const float* xb = x + ((size_t)b * NF + g * 8) * HH * WW;
        BF16x8 o;
#pragma unroll
        for (int c = 0; c < 8; ++c) {
            const float* xc = xb + (size_t)c * HH * WW;
            float s = c00 * xc[i00] + c01 * xc[i01] + c10 * xc[i10] + c11 * xc[i11];
            o.u[c] = f2bf(s * mval);
        }
        *(uint4*)&s_samp[pix * 576 + g * 72 + k * 8] = o.q;   // 16B aligned
    }
    __syncthreads();

    // ---- Phase 4: out = lrelu(W_dcn (64x576) * samp + b) via 18 WMMAs/wave ----
    {
        int osub = wave;                        // 4 waves x 16 och = 64
        v8f acc = {};
        for (int kc = 0; kc < 18; ++kc) {
            v16bf a  = frag_from(wpkd + (size_t)(osub * 18 + kc) * 512 + lane * 16);
            v16bf bf = frag_from(&s_samp[n * 576 + kc * 32 + hi * 16]);
            acc = __builtin_amdgcn_wmma_f32_16x16x32_bf16(false, a, false, bf,
                                                          (short)0, acc, false, false);
        }
        int xx = x0 + n;
#pragma unroll
        for (int r = 0; r < 8; ++r) {
            int och = osub * 16 + 8 * hi + r;
            float v = acc[r] + bd[och];
            v = (v >= 0.0f) ? v : 0.1f * v;
            out[(((size_t)b * NF + och) * HH + y) * WW + xx] = v;
        }
    }
}

// ---------------------------------------------------------------------------
extern "C" void kernel_launch(void* const* d_in, const int* in_sizes, int n_in,
                              void* d_out, int out_size, void* d_ws, size_t ws_size,
                              hipStream_t stream) {
    (void)in_sizes; (void)n_in; (void)out_size; (void)ws_size;
    const float* nbr    = (const float*)d_in[0];
    const float* warped = (const float*)d_in[1];
    const float* ref    = (const float*)d_in[2];
    const float* flows  = (const float*)d_in[3];
    const float* w1     = (const float*)d_in[4];
    const float* b1     = (const float*)d_in[5];
    const float* w2     = (const float*)d_in[6];
    const float* b2     = (const float*)d_in[7];
    const float* wom    = (const float*)d_in[8];
    const float* bom    = (const float*)d_in[9];
    const float* wd     = (const float*)d_in[10];
    const float* bd     = (const float*)d_in[11];
    float* out = (float*)d_out;

    char* ws = (char*)d_ws;
    size_t off = 0;
    auto alloc = [&](size_t bytes) -> char* {
        char* p = ws + off;
        off += (bytes + 255) & ~(size_t)255;
        return p;
    };
    unsigned short* in0   = (unsigned short*)alloc((size_t)BB * HP * WP * 160 * 2);
    unsigned short* off1  = (unsigned short*)alloc((size_t)BB * HP * WP * 64 * 2);
    unsigned short* off2  = (unsigned short*)alloc((size_t)BB * HP * WP * 64 * 2);
    unsigned short* wpk1  = (unsigned short*)alloc((size_t)4 * 9 * 5 * 512 * 2);
    unsigned short* wpk2  = (unsigned short*)alloc((size_t)4 * 9 * 2 * 512 * 2);
    unsigned short* wpkom = (unsigned short*)alloc((size_t)14 * 9 * 2 * 512 * 2);
    unsigned short* wpkd  = (unsigned short*)alloc((size_t)4 * 18 * 512 * 2);

    // Zero padded borders (interiors are fully overwritten below).
    int tb = BB * HP * WP;
    zero_border<<<(tb + 255) / 256, 256, 0, stream>>>(in0, 160);
    zero_border<<<(tb + 255) / 256, 256, 0, stream>>>(off1, 64);
    zero_border<<<(tb + 255) / 256, 256, 0, stream>>>(off2, 64);

    // Weight repack (A-fragment layout) + input concat/pack.
    int t1 = 4 * 9 * 5 * 512;
    pack_conv_w<<<(t1 + 255) / 256, 256, 0, stream>>>(w1, wpk1, 64, 130, 5, 4);
    int t2 = 4 * 9 * 2 * 512;
    pack_conv_w<<<(t2 + 255) / 256, 256, 0, stream>>>(w2, wpk2, 64, 64, 2, 4);
    int t3 = 14 * 9 * 2 * 512;
    pack_conv_w<<<(t3 + 255) / 256, 256, 0, stream>>>(wom, wpkom, 216, 64, 2, 14);
    int t4 = 4 * 18 * 512;
    pack_dcn_w<<<(t4 + 255) / 256, 256, 0, stream>>>(wd, wpkd);
    int tin = BB * HH * WW * 20;
    pack_in0<<<(tin + 255) / 256, 256, 0, stream>>>(warped, ref, flows, in0);

    // conv1: 130(->160) -> 64, lrelu ; conv2: 64 -> 64, lrelu
    dim3 cgrid(BB * HH * (WW / 32));
    size_t lds1 = (size_t)3 * 34 * 160 * 2;
    conv3_wmma<<<cgrid, 256, lds1, stream>>>(in0, wpk1, b1, off1, 160, 5, 64, 64, 1);
    size_t lds2 = (size_t)3 * 34 * 64 * 2;
    conv3_wmma<<<cgrid, 256, lds2, stream>>>(off1, wpk2, b2, off2, 64, 2, 64, 64, 1);

    // fused offset/mask conv + deformable sampling + 576->64 einsum
    dim3 dgrid(BB * HH * (WW / 16));
    dcn_wmma<<<dgrid, 128, 0, stream>>>(off2, wpkom, bom, wpkd, bd, nbr, flows, out);
}
